// JassDNN_69758858822216
// MI455X (gfx1250) — matrix operand order repaired and checked by reading, not compiled, not run on hardware
//
#include <hip/hip_runtime.h>
#include <hip/hip_bf16.h>
#include <math.h>

typedef __attribute__((ext_vector_type(16))) _Float16 v16h;
typedef __attribute__((ext_vector_type(8)))  _Float16 v8h;
typedef __attribute__((ext_vector_type(2)))  _Float16 v2h;
typedef __attribute__((ext_vector_type(8)))  float    v8f;

#define BATCH     16384
#define NUM_CARDS 71
#define NUM_SLOTS 71
#define D0        512
#define D1        256
#define D2        128
#define ACTIONS   9

// ---------------------------------------------------------------------------
// Kernel 0: convert W0 (5048 x 512 f32) to f16 once per launch. Halves the
// dominant L2 gather traffic of the whole network (2.38 GB -> 1.19 GB).
// ---------------------------------------------------------------------------
__global__ void jass_cvt_w0(const float* __restrict__ W0,
                            _Float16* __restrict__ W0h, int n) {
    const int tid = blockIdx.x * blockDim.x + threadIdx.x;
    if (tid < n) W0h[tid] = (_Float16)W0[tid];
}

// ---------------------------------------------------------------------------
// Kernel 1: layer 0 as embedding gather-sum over the f16 W0 copy.
// One block per batch row; 256 threads each own 2 adjacent columns via one
// packed 4-byte load per gathered row. f32 accumulate, ReLU, f16 out.
// ---------------------------------------------------------------------------
__global__ void jass_embed_l0(const int* __restrict__ idx,
                              const float* __restrict__ trump,
                              const _Float16* __restrict__ W0h,
                              const float* __restrict__ b0,
                              _Float16* __restrict__ H0) {
    const int row = blockIdx.x;
    const int c0 = threadIdx.x * 2;
    float acc0 = 0.f, acc1 = 0.f;
    const int* irow = idx + (size_t)row * NUM_SLOTS;
    #pragma unroll 4
    for (int s = 0; s < NUM_SLOTS; ++s) {
        int ci = irow[s];
        ci = ci < 0 ? 0 : (ci > NUM_CARDS - 1 ? NUM_CARDS - 1 : ci);
        const v2h w = *(const v2h*)(W0h + (size_t)(s * NUM_CARDS + ci) * D0 + c0);
        acc0 += (float)w[0];
        acc1 += (float)w[1];
    }
    const float* trow = trump + (size_t)row * 7;
    #pragma unroll
    for (int j = 0; j < 7; ++j) {
        const float tv = trow[j];
        const v2h w = *(const v2h*)(W0h + (size_t)(NUM_SLOTS * NUM_CARDS + j) * D0 + c0);
        acc0 += tv * (float)w[0];
        acc1 += tv * (float)w[1];
    }
    acc0 += b0[c0];
    acc1 += b0[c0 + 1];
    v2h out;
    out[0] = (_Float16)fmaxf(acc0, 0.f);
    out[1] = (_Float16)fmaxf(acc1, 0.f);
    *(v2h*)(H0 + (size_t)row * D0 + c0) = out;
}

// ---------------------------------------------------------------------------
// Kernel 2: pack a row-major f32 weight matrix W (K x N) into the per-lane
// WMMA B-operand layout for v_wmma_f32_16x16x32_f16, converting to f16.
// Layout: Bp[((n_tile*Kb + kb)*32 + lane)*16 + i]
//   lane<16 : column n = n_tile*16+lane,    K = kb*32 + {0..7,16..23}  (i)
//   lane>=16: column n = n_tile*16+lane-16, K = kb*32 + {8..15,24..31} (i)
// ---------------------------------------------------------------------------
__global__ void jass_pack_b(const float* __restrict__ W,
                            _Float16* __restrict__ Bp,
                            int K, int N) {
    const int tid = blockIdx.x * blockDim.x + threadIdx.x;
    if (tid >= K * N) return;
    const int Kb = K >> 5;
    const int i = tid & 15;
    const int lane = (tid >> 4) & 31;
    const int rest = tid >> 9;
    const int kb = rest % Kb;
    const int n_tile = rest / Kb;
    const int n = n_tile * 16 + (lane & 15);
    const int k = kb * 32 + ((lane < 16) ? 0 : 8) + (i & 7) + ((i >> 3) << 4);
    Bp[tid] = (_Float16)W[(size_t)k * N + n];
}

// ---------------------------------------------------------------------------
// Kernel 3: GEMM + bias + ReLU via v_wmma_f32_16x16x32_f16.
// 8 waves/block; each wave computes one 16-row tile x TWO 16-col tiles, so
// every A fragment feeds two WMMAs. KB (K/32) is a compile-time template
// parameter so the K loop fully unrolls; explicit next-step prefetch plus
// __launch_bounds__(256,1) gives the allocator room to double-buffer the two
// B streams in separate registers (avoid s_wait_loadcnt 0x0 full drains).
// A-operand per ISA: lane<16 holds M=lane, K {0..7,16..23}; lane>=16 holds
// M=lane-16, K {8..15,24..31} -> two contiguous 16B loads per K-step.
// ---------------------------------------------------------------------------
template <int KB, int N>
__global__ __launch_bounds__(256, 1)
void jass_gemm_relu_wmma(const _Float16* __restrict__ A,
                         const _Float16* __restrict__ Bp,
                         const float* __restrict__ bias,
                         _Float16* __restrict__ C) {
    const int lane = threadIdx.x & 31;
    const int wave = threadIdx.x >> 5;
    const int m_tile = blockIdx.x * 8 + wave;
    const int n_tile0 = blockIdx.y * 2;
    const int K = KB * 32;

    const int m_lane = lane & 15;
    const int k0 = (lane < 16) ? 0 : 8;

    const _Float16* a_base  = A + (size_t)(m_tile * 16 + m_lane) * K + k0;
    const _Float16* b_base0 = Bp + ((size_t)n_tile0 * KB * 32 + lane) * 16;
    const _Float16* b_base1 = b_base0 + (size_t)KB * 32 * 16;

    auto load_a = [&](int kb) {
        v8h alo = *(const v8h*)(a_base + kb * 32);
        v8h ahi = *(const v8h*)(a_base + kb * 32 + 16);
        v16h a;
        #pragma unroll
        for (int i = 0; i < 8; ++i) { a[i] = alo[i]; a[8 + i] = ahi[i]; }
        return a;
    };

    v8f acc0 = {};
    v8f acc1 = {};

    // Software-pipelined, fully unrolled K loop.
    v16h a_cur  = load_a(0);
    v16h b0_cur = *(const v16h*)(b_base0);
    v16h b1_cur = *(const v16h*)(b_base1);
    #pragma unroll
    for (int kb = 0; kb < KB; ++kb) {
        v16h a_nxt, b0_nxt, b1_nxt;
        if (kb + 1 < KB) {
            a_nxt  = load_a(kb + 1);
            b0_nxt = *(const v16h*)(b_base0 + (size_t)(kb + 1) * 32 * 16);
            b1_nxt = *(const v16h*)(b_base1 + (size_t)(kb + 1) * 32 * 16);
        }
        // (neg_a, A, neg_b, B, c_mod, C, reuse_a, reuse_b)
        acc0 = __builtin_amdgcn_wmma_f32_16x16x32_f16(
            false, a_cur, false, b0_cur, (short)0, acc0, false, false);
        acc1 = __builtin_amdgcn_wmma_f32_16x16x32_f16(
            false, a_cur, false, b1_cur, (short)0, acc1, false, false);
        a_cur = a_nxt; b0_cur = b0_nxt; b1_cur = b1_nxt;
    }

    const int row0 = m_tile * 16 + ((lane < 16) ? 0 : 8);
    #pragma unroll
    for (int half = 0; half < 2; ++half) {
        const int n = (n_tile0 + half) * 16 + m_lane;
        const float bn = bias[n];
        const v8f& acc = half ? acc1 : acc0;
        #pragma unroll
        for (int r = 0; r < 8; ++r) {
            float v = acc[r] + bn;
            v = v > 0.f ? v : 0.f;
            C[(size_t)(row0 + r) * N + n] = (_Float16)v;
        }
    }
}

// ---------------------------------------------------------------------------
// Kernel 4: head (128 -> 9) + legal mask + log-softmax. One thread per row;
// 1152 FMAs/row, Wp (4.6 KB) L2/L0 resident.
// ---------------------------------------------------------------------------
__global__ void jass_head_logsoftmax(const _Float16* __restrict__ H2,
                                     const float* __restrict__ Wp,
                                     const float* __restrict__ bp,
                                     const unsigned char* __restrict__ mask,
                                     float* __restrict__ out) {
    const int row = blockIdx.x * blockDim.x + threadIdx.x;
    if (row >= BATCH) return;

    float logits[ACTIONS];
    #pragma unroll
    for (int a = 0; a < ACTIONS; ++a) logits[a] = bp[a];

    const _Float16* h = H2 + (size_t)row * D2;
    for (int k = 0; k < D2; ++k) {
        const float hv = (float)h[k];
        const float* wrow = Wp + (size_t)k * ACTIONS;
        #pragma unroll
        for (int a = 0; a < ACTIONS; ++a) logits[a] += hv * wrow[a];
    }

    const unsigned char* mrow = mask + (size_t)row * ACTIONS;
    float mx = -INFINITY;
    #pragma unroll
    for (int a = 0; a < ACTIONS; ++a) {
        if (!mrow[a]) logits[a] = -INFINITY;
        mx = fmaxf(mx, logits[a]);
    }
    float s = 0.f;
    #pragma unroll
    for (int a = 0; a < ACTIONS; ++a) s += __expf(logits[a] - mx);
    const float lse = mx + __logf(s);
    float* orow = out + (size_t)row * ACTIONS;
    #pragma unroll
    for (int a = 0; a < ACTIONS; ++a) orow[a] = logits[a] - lse;
}

// ---------------------------------------------------------------------------
// Launcher. Input order (setup_inputs dict order):
//  0 state_idx (B*71 i32), 1 trump (B*7 f32), 2 legal_mask (B*9 bool),
//  3 W0 (5048*512), 4 b0 (512), 5 W1 (512*256), 6 b1 (256),
//  7 W2 (256*128), 8 b2 (128), 9 Wp (128*9), 10 bp (9)
// Output: (B*9) f32 log-probs.
// ---------------------------------------------------------------------------
extern "C" void kernel_launch(void* const* d_in, const int* in_sizes, int n_in,
                              void* d_out, int out_size, void* d_ws, size_t ws_size,
                              hipStream_t stream) {
    const int*   state_idx = (const int*)  d_in[0];
    const float* trump     = (const float*)d_in[1];
    const unsigned char* legal = (const unsigned char*)d_in[2];
    const float* W0 = (const float*)d_in[3];
    const float* b0 = (const float*)d_in[4];
    const float* W1 = (const float*)d_in[5];
    const float* b1 = (const float*)d_in[6];
    const float* W2 = (const float*)d_in[7];
    const float* b2 = (const float*)d_in[8];
    const float* Wp = (const float*)d_in[9];
    const float* bp = (const float*)d_in[10];
    float* out = (float*)d_out;

    const int W0_IN = (NUM_SLOTS * NUM_CARDS + 7);   // 5048 rows

    // Workspace carve-up (all f16), 256B aligned.
    char* ws = (char*)d_ws;
    size_t off = 0;
    auto carve = [&](size_t bytes) {
        char* p = ws + off;
        off += (bytes + 255) & ~(size_t)255;
        return p;
    };
    _Float16* H0  = (_Float16*)carve((size_t)BATCH * D0 * 2);   // 16 MB
    _Float16* H1  = (_Float16*)carve((size_t)BATCH * D1 * 2);   //  8 MB
    _Float16* H2  = (_Float16*)carve((size_t)BATCH * D2 * 2);   //  4 MB
    _Float16* W0h = (_Float16*)carve((size_t)W0_IN * D0 * 2);   // 5.2 MB
    _Float16* W1p = (_Float16*)carve((size_t)D0 * D1 * 2);      // 256 KB
    _Float16* W2p = (_Float16*)carve((size_t)D1 * D2 * 2);      //  64 KB

    // Weight conversions / repacks (deterministic, run every call).
    {
        int n = W0_IN * D0;
        jass_cvt_w0<<<(n + 255) / 256, 256, 0, stream>>>(W0, W0h, n);
    }
    {
        int n = D0 * D1;
        jass_pack_b<<<(n + 255) / 256, 256, 0, stream>>>(W1, W1p, D0, D1);
    }
    {
        int n = D1 * D2;
        jass_pack_b<<<(n + 255) / 256, 256, 0, stream>>>(W2, W2p, D1, D2);
    }

    // Layer 0: embedding gather-sum + trump tail + bias + ReLU (f16 gathers).
    jass_embed_l0<<<BATCH, 256, 0, stream>>>(state_idx, trump, W0h, b0, H0);

    // Layer 1: (16384 x 512) x (512 x 256) WMMA GEMM + ReLU.
    {
        dim3 grid(BATCH / 128, D1 / 32);
        jass_gemm_relu_wmma<16, D1><<<grid, 256, 0, stream>>>(H0, W1p, b1, H1);
    }
    // Layer 2: (16384 x 256) x (256 x 128) WMMA GEMM + ReLU.
    {
        dim3 grid(BATCH / 128, D2 / 32);
        jass_gemm_relu_wmma<8, D2><<<grid, 256, 0, stream>>>(H1, W2p, b2, H2);
    }
    // Head + log-softmax.
    jass_head_logsoftmax<<<BATCH / 256, 256, 0, stream>>>(H2, Wp, bp, legal, out);
}
